// HGATLinkConv_84980222919196
// MI455X (gfx1250) — compile-verified, matrix operand order checked by reading, not run on hardware
//
#include <hip/hip_runtime.h>
#include <hip/hip_bf16.h>

typedef __bf16 bf16_t;
typedef bf16_t v16bf __attribute__((ext_vector_type(16)));
typedef bf16_t v8bf  __attribute__((ext_vector_type(8)));
typedef float  v8f   __attribute__((ext_vector_type(8)));

#define N_NODES 10000
#define IN_FEAT 256
#define OUT_FEAT 128
#define N_EDGES 640000
#define INV_TAU 4.0f

// ---------------- prep: f32 -> bf16 feat ----------------
__global__ void prep_feat_kernel(const float* __restrict__ F, bf16_t* __restrict__ Fb) {
    int t = blockIdx.x * blockDim.x + threadIdx.x;           // one float4 each
    if (t < N_NODES * IN_FEAT / 4) {
        float4 v = ((const float4*)F)[t];
        bf16_t* o = Fb + (size_t)t * 4;
        o[0] = (bf16_t)v.x; o[1] = (bf16_t)v.y; o[2] = (bf16_t)v.z; o[3] = (bf16_t)v.w;
    }
}

// ---------------- prep: transpose + convert both weights ----------------
// W row-major [IN_FEAT][OUT_FEAT]; Wt[n][k] = W[k][n], bf16
__global__ void prep_weights_kernel(const float* __restrict__ W, const float* __restrict__ Wk,
                                    bf16_t* __restrict__ Wt, bf16_t* __restrict__ Wkt) {
    int t = blockIdx.x * blockDim.x + threadIdx.x;
    if (t < OUT_FEAT * IN_FEAT) {
        int n = t / IN_FEAT;
        int k = t - n * IN_FEAT;
        Wt[t]  = (bf16_t)W[(size_t)k * OUT_FEAT + n];
        Wkt[t] = (bf16_t)Wk[(size_t)k * OUT_FEAT + n];
    }
}

// ---------------- prep: zero the scatter-max target ----------------
__global__ void zero_rst_kernel(uint4* __restrict__ rst) {
    int t = blockIdx.x * blockDim.x + threadIdx.x;
    if (t < N_NODES * OUT_FEAT / 4) {
        uint4 z; z.x = 0u; z.y = 0u; z.z = 0u; z.w = 0u;
        rst[t] = z;
    }
}

// ---------------- fused dual GEMM: h = relu((feat@W)*cj), q = (feat@Wk)*ci ----------------
// One wave computes one 16x16 tile of both outputs. grid = (8 ntiles, 625 mtiles), block = 32.
__global__ void gemm_dual_kernel(const bf16_t* __restrict__ A,    // [N][256] bf16
                                 const bf16_t* __restrict__ Wt,   // [128][256] bf16 (W^T)
                                 const bf16_t* __restrict__ Wkt,  // [128][256] bf16 (Wk^T)
                                 const float*  __restrict__ ci,
                                 const float*  __restrict__ cj,
                                 float* __restrict__ Hout,        // [N][128]
                                 float* __restrict__ Qout) {      // [N][128]
    const int ntile = blockIdx.x;          // 0..7
    const int mtile = blockIdx.y;          // 0..624
    const int lane  = threadIdx.x;         // 0..31 (wave32)
    const int m     = lane & 15;
    const int kg    = lane >> 4;           // K-group select per ISA A/B layout

    const bf16_t* aRow  = A   + (size_t)(mtile * 16 + m) * IN_FEAT;
    const bf16_t* bRow  = Wt  + (size_t)(ntile * 16 + m) * IN_FEAT;
    const bf16_t* bkRow = Wkt + (size_t)(ntile * 16 + m) * IN_FEAT;

    v8f accH = {};
    v8f accQ = {};

#pragma unroll
    for (int k0 = 0; k0 < IN_FEAT; k0 += 32) {
        const int kOff = k0 + kg * 8;
        // A fragment: rows striped per lane, K runs [kOff, kOff+8) and [kOff+16, kOff+24)
        v8bf alo = *(const v8bf*)(aRow + kOff);
        v8bf ahi = *(const v8bf*)(aRow + kOff + 16);
        v16bf a  = __builtin_shufflevector(alo, ahi, 0,1,2,3,4,5,6,7,8,9,10,11,12,13,14,15);
        // B fragment: column n = lane&15 of W == row of Wt, same K pattern
        v8bf blo = *(const v8bf*)(bRow + kOff);
        v8bf bhi = *(const v8bf*)(bRow + kOff + 16);
        v16bf b  = __builtin_shufflevector(blo, bhi, 0,1,2,3,4,5,6,7,8,9,10,11,12,13,14,15);
        v8bf klo = *(const v8bf*)(bkRow + kOff);
        v8bf khi = *(const v8bf*)(bkRow + kOff + 16);
        v16bf bk = __builtin_shufflevector(klo, khi, 0,1,2,3,4,5,6,7,8,9,10,11,12,13,14,15);

        accH = __builtin_amdgcn_wmma_f32_16x16x32_bf16(false, a, false, b,  (short)0, accH, false, false);
        accQ = __builtin_amdgcn_wmma_f32_16x16x32_bf16(false, a, false, bk, (short)0, accQ, false, false);
    }

    // C/D layout: VGPR i holds row (lane>>4)*8 + i, col lane&15
    const int col   = ntile * 16 + (lane & 15);
    const int rbase = mtile * 16 + (lane >> 4) * 8;
#pragma unroll
    for (int i = 0; i < 8; ++i) {
        const int row = rbase + i;
        const float cjv = cj[row];
        const float civ = ci[row];
        float h = accH[i] * cjv;
        h = h > 0.0f ? h : 0.0f;                       // relu
        Hout[(size_t)row * OUT_FEAT + col] = h;
        Qout[(size_t)row * OUT_FEAT + col] = accQ[i] * civ;
    }
}

// ---------------- attn: per-row head-normalize + softmax (in place on Q) ----------------
// One wave per row: 32 lanes x float4 = 128 cols. Head = 16 cols = 4 lanes.
__global__ void attn_kernel(float* __restrict__ Q) {
    const int row  = blockIdx.x * (blockDim.x >> 5) + (threadIdx.x >> 5);
    const int lane = threadIdx.x & 31;
    if (row >= N_NODES) return;

    float4 q = ((const float4*)(Q + (size_t)row * OUT_FEAT))[lane];

    // per-head sum of squares (4 lanes per head)
    float ss = q.x * q.x + q.y * q.y + q.z * q.z + q.w * q.w;
    ss += __shfl_xor(ss, 1, 32);
    ss += __shfl_xor(ss, 2, 32);
    // alpha = (q/max(||q||,eps))^2 = q^2 / max(ss, eps^2)
    const float inv = 1.0f / fmaxf(ss, 1e-24f);
    float a0 = q.x * q.x * inv;
    float a1 = q.y * q.y * inv;
    float a2 = q.z * q.z * inv;
    float a3 = q.w * q.w * inv;

    // softmax(alpha / TAU) over the 128-wide row
    float mx = fmaxf(fmaxf(a0, a1), fmaxf(a2, a3));
    mx = fmaxf(mx, __shfl_xor(mx, 1, 32));
    mx = fmaxf(mx, __shfl_xor(mx, 2, 32));
    mx = fmaxf(mx, __shfl_xor(mx, 4, 32));
    mx = fmaxf(mx, __shfl_xor(mx, 8, 32));
    mx = fmaxf(mx, __shfl_xor(mx, 16, 32));

    float e0 = __expf((a0 - mx) * INV_TAU);
    float e1 = __expf((a1 - mx) * INV_TAU);
    float e2 = __expf((a2 - mx) * INV_TAU);
    float e3 = __expf((a3 - mx) * INV_TAU);

    float sum = e0 + e1 + e2 + e3;
    sum += __shfl_xor(sum, 1, 32);
    sum += __shfl_xor(sum, 2, 32);
    sum += __shfl_xor(sum, 4, 32);
    sum += __shfl_xor(sum, 8, 32);
    sum += __shfl_xor(sum, 16, 32);
    const float r = 1.0f / sum;

    float4 o; o.x = e0 * r; o.y = e1 * r; o.z = e2 * r; o.w = e3 * r;
    ((float4*)(Q + (size_t)row * OUT_FEAT))[lane] = o;
}

// ---------------- scatter-max over edges (u32 atomic max == f32 max for x>=0) ----------------
// 32 lanes per edge, float4 per lane. h rows sit in L2 (5 MB << 192 MB).
__global__ void scatter_max_kernel(const int* __restrict__ src, const int* __restrict__ dst,
                                   const float* __restrict__ H, unsigned int* __restrict__ rst) {
    const int t    = blockIdx.x * blockDim.x + threadIdx.x;
    const int e    = t >> 5;
    const int lane = t & 31;
    if (e >= N_EDGES) return;
    const int s = src[e];
    const int d = dst[e];
    float4 v = ((const float4*)(H + (size_t)s * OUT_FEAT))[lane];
    unsigned int* o = rst + (size_t)d * OUT_FEAT + lane * 4;
    if (v.x != 0.0f) atomicMax(o + 0, __float_as_uint(v.x));
    if (v.y != 0.0f) atomicMax(o + 1, __float_as_uint(v.y));
    if (v.z != 0.0f) atomicMax(o + 2, __float_as_uint(v.z));
    if (v.w != 0.0f) atomicMax(o + 3, __float_as_uint(v.w));
}

// ---------------- out = rst * attn ----------------
__global__ void finalize_kernel(const float* __restrict__ rst, const float* __restrict__ attn,
                                float* __restrict__ out) {
    int t = blockIdx.x * blockDim.x + threadIdx.x;
    if (t < N_NODES * OUT_FEAT / 4) {
        float4 r = ((const float4*)rst)[t];
        float4 a = ((const float4*)attn)[t];
        float4 o; o.x = r.x * a.x; o.y = r.y * a.y; o.z = r.z * a.z; o.w = r.w * a.w;
        ((float4*)out)[t] = o;
    }
}

extern "C" void kernel_launch(void* const* d_in, const int* in_sizes, int n_in,
                              void* d_out, int out_size, void* d_ws, size_t ws_size,
                              hipStream_t stream) {
    const float* feat   = (const float*)d_in[0];   // [10000,256]
    const float* ci     = (const float*)d_in[1];   // [10000,1]
    const float* cj     = (const float*)d_in[2];   // [10000,1]
    const float* weight = (const float*)d_in[3];   // [256,128]
    const float* wk     = (const float*)d_in[4];   // [256,128]
    const int*   src    = (const int*)d_in[5];     // [640000]
    const int*   dst    = (const int*)d_in[6];     // [640000]
    float* out = (float*)d_out;                    // [10000,128]

    // workspace layout (all offsets 256B-aligned)
    char* ws = (char*)d_ws;
    const size_t szFeatB = (size_t)N_NODES * IN_FEAT * sizeof(bf16_t);    // 5,120,000
    const size_t szWtB   = (size_t)OUT_FEAT * IN_FEAT * sizeof(bf16_t);   // 65,536
    const size_t szF32   = (size_t)N_NODES * OUT_FEAT * sizeof(float);    // 5,120,000
    bf16_t* featb = (bf16_t*)(ws);
    bf16_t* Wt    = (bf16_t*)(ws + szFeatB);
    bf16_t* Wkt   = (bf16_t*)(ws + szFeatB + szWtB);
    float*  H     = (float*) (ws + szFeatB + 2 * szWtB);
    float*  Q     = (float*) (ws + szFeatB + 2 * szWtB + szF32);          // q then attn (in place)
    unsigned int* rst = (unsigned int*)(ws + szFeatB + 2 * szWtB + 2 * szF32);

    // 1) conversions / init (independent; same stream keeps order vs consumers)
    prep_feat_kernel<<<(N_NODES * IN_FEAT / 4 + 255) / 256, 256, 0, stream>>>(feat, featb);
    prep_weights_kernel<<<(OUT_FEAT * IN_FEAT + 255) / 256, 256, 0, stream>>>(weight, wk, Wt, Wkt);
    zero_rst_kernel<<<(N_NODES * OUT_FEAT / 4 + 255) / 256, 256, 0, stream>>>((uint4*)rst);

    // 2) fused dual WMMA GEMM + epilogue
    dim3 ggrid(OUT_FEAT / 16, N_NODES / 16);   // (8, 625)
    gemm_dual_kernel<<<ggrid, 32, 0, stream>>>(featb, Wt, Wkt, ci, cj, H, Q);

    // 3) head-norm + softmax (in place on Q)
    attn_kernel<<<(N_NODES + 7) / 8, 256, 0, stream>>>(Q);

    // 4) edge scatter-max
    scatter_max_kernel<<<(N_EDGES * 32 + 255) / 256, 256, 0, stream>>>(src, dst, H, rst);

    // 5) out = rst * attn
    finalize_kernel<<<(N_NODES * OUT_FEAT / 4 + 255) / 256, 256, 0, stream>>>((const float*)rst, Q, out);
}
